// CMambaEncoder_32598801776663
// MI455X (gfx1250) — matrix operand order repaired and checked by reading, not compiled, not run on hardware
//
#include <hip/hip_runtime.h>
#include <hip/hip_bf16.h>

typedef __attribute__((ext_vector_type(16))) _Float16 v16h;
typedef __attribute__((ext_vector_type(8)))  _Float16 v8h;
typedef __attribute__((ext_vector_type(8)))  float    v8f;

#define LSTR 72   // padded LDS row stride in halves for BK=64 (144B = 9x16B; 36*r mod 64 distinct -> conflict-free)

// Split workgroup barrier with LDS-only drain: do NOT wait for in-flight global
// prefetch loads (VGPR-private), only for our ds_stores to be visible (DScnt=0).
// __syncthreads() would drain LOADcnt too and destroy the prefetch pipeline.
__device__ __forceinline__ void lds_barrier() {
    asm volatile("s_wait_dscnt 0x0\n\t"
                 "s_barrier_signal -1\n\t"
                 "s_barrier_wait -1" ::: "memory");
}

// ---------- WMMA fragment loaders (wave32, 16x16x32 f16) ----------
__device__ __forceinline__ v16h fragA(const _Float16* t, int lane) {
    // A 16x32 (MxK): lane<16 row M=lane holds K 0..7,16..23 ; lane>=16 row M=lane-16 holds K 8..15,24..31
    int r  = lane & 15;
    int c0 = (lane & 16) ? 8 : 0;
    const _Float16* p = t + r * LSTR + c0;
    v8h lo = *(const v8h*)p;
    v8h hi = *(const v8h*)(p + 16);
    v16h o;
#pragma unroll
    for (int i = 0; i < 8; i++) { o[i] = lo[i]; o[i + 8] = hi[i]; }
    return o;
}
__device__ __forceinline__ v16h fragB(const _Float16* t, int lane) {
    // B 32x16 (KxN), stored in LDS as N-major rows (weights are [N][K] row-major):
    // lane<16 col N=lane holds K 0..7,16..23 ; lane>=16 col N=lane-16 holds K 8..15,24..31
    int c  = lane & 15;
    int k0 = (lane & 16) ? 8 : 0;
    const _Float16* p = t + c * LSTR + k0;
    v8h lo = *(const v8h*)p;
    v8h hi = *(const v8h*)(p + 16);
    v16h o;
#pragma unroll
    for (int i = 0; i < 8; i++) { o[i] = lo[i]; o[i + 8] = hi[i]; }
    return o;
}

__device__ __forceinline__ float silu_f(float v)   { return v / (1.f + __expf(-v)); }
__device__ __forceinline__ float sigm_f(float v)   { return 1.f / (1.f + __expf(-v)); }
__device__ __forceinline__ float gelu_f(float v)   { return 0.5f * v * (1.f + erff(v * 0.70710678118f)); }
__device__ __forceinline__ float wred_sum(float v) {
#pragma unroll
    for (int off = 16; off; off >>= 1) v += __shfl_xor(v, off, 32);
    return v;
}
__device__ __forceinline__ float wred_max(float v) {
#pragma unroll
    for (int off = 16; off; off >>= 1) v = fmaxf(v, __shfl_xor(v, off, 32));
    return v;
}

// ---------- weight f32 -> f16 conversion (x_proj padded 296 -> 320 rows) ----------
__global__ void cmamba_convert_w(const float* __restrict__ inw, const float* __restrict__ xpw,
                                 const float* __restrict__ outw,
                                 _Float16* __restrict__ inw16, _Float16* __restrict__ xpw16,
                                 _Float16* __restrict__ outw16) {
    const int S1 = 2 * 512 * 128;       // in_proj
    const int S2 = 2 * 320 * 256;       // x_proj padded
    const int S3 = 2 * 128 * 256;       // out_proj
    int i = blockIdx.x * blockDim.x + threadIdx.x;
    if (i < S1) {
        inw16[i] = (_Float16)inw[i];
    } else if (i < S1 + S2) {
        int j = i - S1;
        int e = j / (320 * 256);
        int r = (j / 256) % 320;
        int c = j % 256;
        xpw16[j] = (r < 296) ? (_Float16)xpw[((size_t)e * 296 + r) * 256 + c] : (_Float16)0.f;
    } else if (i < S1 + S2 + S3) {
        int j = i - S1 - S2;
        outw16[j] = (_Float16)outw[j];
    }
}

// ---------- gddMLP + rmsnorm fused: one block per (b,p), all 16 n-rows ----------
__global__ __launch_bounds__(128) void cmamba_gdd(const float* __restrict__ h,
                                                  const float* __restrict__ sc1, const float* __restrict__ sc2,
                                                  const float* __restrict__ sf1, const float* __restrict__ sf2,
                                                  const float* __restrict__ normw,
                                                  float* __restrict__ g, _Float16* __restrict__ xn16) {
    __shared__ float tile[16][128];
    __shared__ float avgN[16], mxN[16], gsc[8], gsf[8], sscale[16], sshift[16];
    const int b = blockIdx.x >> 6, p = blockIdx.x & 63;
    const int t = threadIdx.x, lane = t & 31, w = t >> 5;

    for (int n = 0; n < 16; n++)
        tile[n][t] = h[(((size_t)(b * 16 + n) * 64 + p) * 128) + t];
    __syncthreads();

    for (int n = w; n < 16; n += 4) {
        float sm = 0.f, mx = -3.4e38f;
#pragma unroll
        for (int q = 0; q < 4; q++) {
            float v = tile[n][lane + 32 * q];
            sm += v; mx = fmaxf(mx, v);
        }
        sm = wred_sum(sm); mx = wred_max(mx);
        if (lane == 0) { avgN[n] = sm * (1.f / 128.f); mxN[n] = mx; }
    }
    __syncthreads();

    if (t < 8) {               // scale hidden
        float da = 0.f, dm = 0.f;
#pragma unroll
        for (int i = 0; i < 16; i++) { da += sc1[t * 16 + i] * avgN[i]; dm += sc1[t * 16 + i] * mxN[i]; }
        gsc[t] = gelu_f(da) + gelu_f(dm);
    } else if (t < 16) {       // shift hidden
        int j = t - 8;
        float da = 0.f, dm = 0.f;
#pragma unroll
        for (int i = 0; i < 16; i++) { da += sf1[j * 16 + i] * avgN[i]; dm += sf1[j * 16 + i] * mxN[i]; }
        gsf[j] = gelu_f(da) + gelu_f(dm);
    }
    __syncthreads();

    if (t < 16) {
        float s = 0.f;
#pragma unroll
        for (int j = 0; j < 8; j++) s += sc2[t * 8 + j] * gsc[j];
        sscale[t] = sigm_f(s);
    } else if (t < 32) {
        int n = t - 16; float s = 0.f;
#pragma unroll
        for (int j = 0; j < 8; j++) s += sf2[n * 8 + j] * gsf[j];
        sshift[n] = sigm_f(s);
    }
    __syncthreads();

    for (int n = w; n < 16; n += 4) {
        float gv[4]; float ssq = 0.f;
#pragma unroll
        for (int q = 0; q < 4; q++) {
            float x = tile[n][lane + 32 * q];
            gv[q] = sscale[n] * x + sshift[n];
            ssq += gv[q] * gv[q];
        }
        ssq = wred_sum(ssq);
        float rms = rsqrtf(ssq * (1.f / 128.f) + 1e-5f);
        size_t base = ((size_t)(b * 16 + n) * 64 + p) * 128;
#pragma unroll
        for (int q = 0; q < 4; q++) {
            int d = lane + 32 * q;
            g[base + d]    = gv[q];
            xn16[base + d] = (_Float16)(gv[q] * rms * normw[d]);
        }
    }
}

// ---------- WMMA GEMM: C[M,N] = A[M,K] * B[N,K]^T ----------
// 64x64 tiles, 8 waves (16x32 each), BK=64, ping-pong LDS + register prefetch pipeline.
// MODE 0: A = xn16 (f16);        epilogue: silu + split -> xi32 / zs
// MODE 1: A = xi32 (f32->f16);   epilogue: store f32 (stride N)
// MODE 2: A = y32*zs (f32->f16); epilogue: + residual -> out (stride N)
template <int MODE>
__global__ __launch_bounds__(256) void cmamba_gemm(const _Float16* __restrict__ A16,
                                                   const float* __restrict__ Ay, const float* __restrict__ Az,
                                                   const _Float16* __restrict__ Bw,
                                                   float* __restrict__ out0, float* __restrict__ out1,
                                                   const float* __restrict__ resid,
                                                   int M, int N, int K) {
    __shared__ __align__(16) _Float16 As[2][64 * LSTR];
    __shared__ __align__(16) _Float16 Bs[2][64 * LSTR];
    const int t = threadIdx.x, lane = t & 31, w = t >> 5;
    const int m0 = blockIdx.x * 64, n0 = blockIdx.y * 64;
    const int msub = (w & 3) * 16, nsub = (w >> 2) * 32;
    const int lr = t >> 2, lc0 = (t & 3) * 16;   // staging: row, 16-half column chunk

    const size_t arow = (size_t)(m0 + lr) * K + lc0;
    const size_t brow = (size_t)(n0 + lr) * K + lc0;

    v8h pa0, pa1, pb0, pb1;     // MODE 0 A prefetch + B prefetch
    float4 pay[4], paz[4];      // MODE 1/2 A prefetch (f32)

    auto prefetch = [&](int kk) {
        if (MODE == 0) {
            pa0 = *(const v8h*)(A16 + arow + kk);
            pa1 = *(const v8h*)(A16 + arow + kk + 8);
        } else {
            const float4* py = (const float4*)(Ay + arow + kk);
#pragma unroll
            for (int i = 0; i < 4; i++) pay[i] = py[i];
            if (MODE == 2) {
                const float4* pz = (const float4*)(Az + arow + kk);
#pragma unroll
                for (int i = 0; i < 4; i++) paz[i] = pz[i];
            }
        }
        pb0 = *(const v8h*)(Bw + brow + kk);
        pb1 = *(const v8h*)(Bw + brow + kk + 8);
    };
    auto stage = [&](int buf) {
        _Float16* da = &As[buf][lr * LSTR + lc0];
        if (MODE == 0) {
            *(v8h*)da = pa0; *(v8h*)(da + 8) = pa1;
        } else {
            float f[16];
#pragma unroll
            for (int i = 0; i < 4; i++) {
                f[4*i+0] = pay[i].x; f[4*i+1] = pay[i].y; f[4*i+2] = pay[i].z; f[4*i+3] = pay[i].w;
            }
            if (MODE == 2) {
                float z[16];
#pragma unroll
                for (int i = 0; i < 4; i++) {
                    z[4*i+0] = paz[i].x; z[4*i+1] = paz[i].y; z[4*i+2] = paz[i].z; z[4*i+3] = paz[i].w;
                }
#pragma unroll
                for (int i = 0; i < 16; i++) f[i] *= z[i];
            }
            v8h h0, h1;
#pragma unroll
            for (int i = 0; i < 8; i++) { h0[i] = (_Float16)f[i]; h1[i] = (_Float16)f[i + 8]; }
            *(v8h*)da = h0; *(v8h*)(da + 8) = h1;
        }
        _Float16* db = &Bs[buf][lr * LSTR + lc0];
        *(v8h*)db = pb0; *(v8h*)(db + 8) = pb1;
    };

    v8f c0 = {}; v8f c1 = {};
    prefetch(0);
    const int nst = K >> 6;
    for (int s = 0; s < nst; s++) {
        const int buf = s & 1;
        stage(buf);
        if (s + 1 < nst) prefetch((s + 1) << 6);   // next-stage global loads stay in flight across the barrier
        lds_barrier();                             // DScnt-only drain; ping-pong makes one barrier/stage sufficient
#pragma unroll
        for (int ks = 0; ks < 2; ks++) {
            v16h a  = fragA(&As[buf][msub * LSTR + ks * 32], lane);
            v16h b0 = fragB(&Bs[buf][nsub * LSTR + ks * 32], lane);
            v16h b1 = fragB(&Bs[buf][(nsub + 16) * LSTR + ks * 32], lane);
            c0 = __builtin_amdgcn_wmma_f32_16x16x32_f16(false, a, false, b0, (short)0, c0, false, false);
            c1 = __builtin_amdgcn_wmma_f32_16x16x32_f16(false, a, false, b1, (short)0, c1, false, false);
        }
    }

    const int nc = lane & 15, hi8 = (lane >> 4) * 8;
#pragma unroll
    for (int r = 0; r < 8; r++) {
        int m  = m0 + msub + hi8 + r;
        int na = n0 + nsub + nc;
        int nb = na + 16;
        float va = c0[r], vb = c1[r];
        if (MODE == 0) {
            float sa = silu_f(va), sb = silu_f(vb);
            if (na < 256) out0[(size_t)m * 256 + na] = sa;
            else          out1[(size_t)m * 256 + (na - 256)] = sa;
            if (nb < 256) out0[(size_t)m * 256 + nb] = sb;
            else          out1[(size_t)m * 256 + (nb - 256)] = sb;
        } else if (MODE == 1) {
            out0[(size_t)m * N + na] = va;
            out0[(size_t)m * N + nb] = vb;
        } else {
            out0[(size_t)m * N + na] = va + resid[(size_t)m * N + na];
            out0[(size_t)m * N + nb] = vb + resid[(size_t)m * N + nb];
        }
    }
}

// ---------- fused selective scan: block = one (b*n) sequence, thread = d_ff channel ----------
__global__ __launch_bounds__(256) void cmamba_scan(const float* __restrict__ dbcd,  // [8192][320]
                                                   const float* __restrict__ xi32,  // [8192][256]
                                                   const float* __restrict__ dtw,   // [256][8]
                                                   const float* __restrict__ dtb,   // [256]
                                                   const float* __restrict__ Alog,  // [16]
                                                   float* __restrict__ y32) {       // [8192][256]
    __shared__ float sBC[64 * 40];   // per step: dlt[8] | B[16] | C[16]
    __shared__ float sA[16];
    const int bn = blockIdx.x, t = threadIdx.x;

    for (int i = t; i < 64 * 40; i += 256) {
        int pp = i / 40, c = i % 40;
        sBC[i] = dbcd[((size_t)(bn * 64 + pp)) * 320 + c];
    }
    if (t < 16) sA[t] = -__expf(Alog[t]);
    float w8[8];
#pragma unroll
    for (int i = 0; i < 8; i++) w8[i] = dtw[t * 8 + i];
    const float bt = dtb[t];
    __syncthreads();

    float s[16];
#pragma unroll
    for (int j = 0; j < 16; j++) s[j] = 0.f;

    for (int p = 0; p < 64; p++) {
        const float* row = &sBC[p * 40];     // LDS broadcast reads, conflict-free
        float dot = bt;
#pragma unroll
        for (int i = 0; i < 8; i++) dot += w8[i] * row[i];
        float delta = (dot > 20.f) ? dot : __logf(1.f + __expf(dot));
        size_t m = (size_t)bn * 64 + p;
        float xv = xi32[m * 256 + t];
        float Dv = dbcd[m * 320 + 40 + t];
        float dx = delta * xv;
        float acc = 0.f;
#pragma unroll
        for (int j = 0; j < 16; j++) {
            float dA = __expf(delta * sA[j]);
            s[j] = dA * s[j] + dx * row[8 + j];
            acc += s[j] * row[24 + j];
        }
        y32[m * 256 + t] = acc + Dv * xv;
    }
}

// ---------- final rmsnorm ----------
__global__ __launch_bounds__(128) void cmamba_fnorm(const float* __restrict__ h,
                                                    const float* __restrict__ fnw,
                                                    float* __restrict__ out) {
    __shared__ float part[4];
    const int t = threadIdx.x, lane = t & 31, w = t >> 5;
    const size_t base = (size_t)blockIdx.x * 128;
    float v  = h[base + t];
    float ss = wred_sum(v * v);
    if (lane == 0) part[w] = ss;
    __syncthreads();
    float tot = part[0] + part[1] + part[2] + part[3];
    float rms = rsqrtf(tot * (1.f / 128.f) + 1e-5f);
    out[base + t] = v * rms * fnw[t];
}

extern "C" void kernel_launch(void* const* d_in, const int* in_sizes, int n_in,
                              void* d_out, int out_size, void* d_ws, size_t ws_size,
                              hipStream_t stream) {
    (void)in_sizes; (void)n_in; (void)out_size; (void)ws_size;
    const float* x     = (const float*)d_in[0];
    const float* in_w  = (const float*)d_in[1];
    const float* xp_w  = (const float*)d_in[2];
    const float* dt_w  = (const float*)d_in[3];
    const float* dt_b  = (const float*)d_in[4];
    const float* A_log = (const float*)d_in[5];
    const float* out_w = (const float*)d_in[6];
    const float* normw = (const float*)d_in[7];
    const float* sc1   = (const float*)d_in[8];
    const float* sc2   = (const float*)d_in[9];
    const float* sf1   = (const float*)d_in[10];
    const float* sf2   = (const float*)d_in[11];
    const float* fnw   = (const float*)d_in[12];
    float* out = (float*)d_out;

    char* p = (char*)d_ws;
    auto alloc = [&](size_t bytes) -> char* {
        char* r = p; p += (bytes + 255) & ~(size_t)255; return r;
    };
    float*    g      = (float*)   alloc((size_t)8192 * 128 * 4);
    _Float16* xn16   = (_Float16*)alloc((size_t)8192 * 128 * 2);
    float*    xi32   = (float*)   alloc((size_t)8192 * 256 * 4);
    float*    zs     = (float*)   alloc((size_t)8192 * 256 * 4);
    float*    dbcd   = (float*)   alloc((size_t)8192 * 320 * 4);
    float*    y32    = (float*)   alloc((size_t)8192 * 256 * 4);
    float*    hbuf   = (float*)   alloc((size_t)8192 * 128 * 4);
    _Float16* inw16  = (_Float16*)alloc((size_t)2 * 512 * 128 * 2);
    _Float16* xpw16  = (_Float16*)alloc((size_t)2 * 320 * 256 * 2);
    _Float16* outw16 = (_Float16*)alloc((size_t)2 * 128 * 256 * 2);

    cmamba_convert_w<<<1408, 256, 0, stream>>>(in_w, xp_w, out_w, inw16, xpw16, outw16);

    for (int e = 0; e < 2; e++) {
        const float* hc = e ? (const float*)hbuf : x;
        cmamba_gdd<<<512, 128, 0, stream>>>(hc, sc1 + e * 128, sc2 + e * 128,
                                            sf1 + e * 128, sf2 + e * 128,
                                            normw + e * 128, g, xn16);
        // xz = xn @ in_w^T ; silu-split into xi / silu(z)
        cmamba_gemm<0><<<dim3(128, 8), 256, 0, stream>>>(xn16, nullptr, nullptr,
                                                         inw16 + (size_t)e * 512 * 128,
                                                         xi32, zs, nullptr, 8192, 512, 128);
        // dbcd = xi @ xp_w^T (N padded 296->320)
        cmamba_gemm<1><<<dim3(128, 5), 256, 0, stream>>>(nullptr, xi32, nullptr,
                                                         xpw16 + (size_t)e * 320 * 256,
                                                         dbcd, nullptr, nullptr, 8192, 320, 256);
        cmamba_scan<<<128, 256, 0, stream>>>(dbcd, xi32, dt_w + e * 256 * 8,
                                             dt_b + e * 256, A_log + e * 16, y32);
        // h = (y * silu(z)) @ out_w^T + g
        cmamba_gemm<2><<<dim3(128, 2), 256, 0, stream>>>(nullptr, y32, zs,
                                                         outw16 + (size_t)e * 128 * 256,
                                                         hbuf, nullptr, g, 8192, 128, 256);
    }
    cmamba_fnorm<<<8192, 128, 0, stream>>>(hbuf, fnw, out);
}
